// PointConvD_71863392797313
// MI455X (gfx1250) — matrix operand order repaired and checked by reading, not compiled, not run on hardware
//
#include <hip/hip_runtime.h>
#include <hip/hip_bf16.h>

typedef __attribute__((ext_vector_type(16))) __bf16 v16bf;
typedef __attribute__((ext_vector_type(8)))  __bf16 v8bf;
typedef __attribute__((ext_vector_type(8)))  float  v8f;
typedef __attribute__((ext_vector_type(4)))  unsigned int u32x4;
typedef __attribute__((ext_vector_type(8)))  int    i32x8;
typedef __attribute__((ext_vector_type(4)))  int    i32x4;

#define NB   8
#define NPT  8192
#define DFE  61
#define SS   2048
#define KN   32
#define CIN  64      // 3 + 61
#define WNW  16
#define KDIM 1024    // CIN * WNW
#define OCH  128
#define KNP  48      // padded K-stride for np_s rows (96B, 16B-aligned, de-phased)

// padded LDS image of one 16x1024 bf16 A panel: 8192 dwords + 4 pad dwords
// every 256 dwords (TDM pad_interval=256dw, pad_amount=4dw) = 33280 bytes
#define APAD_BYTES 33280

__device__ __forceinline__ unsigned lds_offset_of(const void* p) {
    // shared-aperture flat address: ADDR[31:0] == LDS byte offset (ISA 10.2)
    return (unsigned)(unsigned long long)p;
}

// ---------------------------------------------------------------------------
// 0) lw [128,1024] f32 -> lwT [1024,128] bf16 (K-major: per k, the 16 lanes' B-frag
//    elements are 32 contiguous bytes)
// ---------------------------------------------------------------------------
__global__ __launch_bounds__(256) void prep_lw_kernel(const float* __restrict__ lw,
                                                      __bf16* __restrict__ lwT) {
    int i = blockIdx.x * blockDim.x + threadIdx.x;   // 0 .. 131071
    if (i < OCH * KDIM) {
        int oc = i >> 10;
        int k  = i & (KDIM - 1);
        lwT[k * OCH + oc] = (__bf16)lw[i];
    }
}

// ---------------------------------------------------------------------------
// 1) Furthest point sampling: one block per batch, min-dist in LDS, serial
//    2048-step loop with block argmax reduction.
// ---------------------------------------------------------------------------
__global__ __launch_bounds__(512) void fps_kernel(const float* __restrict__ xyz, // [B,3,N]
                                                  int* __restrict__ fps_idx,     // [B,S]
                                                  float* __restrict__ qxyz,      // [B,S,3]
                                                  float* __restrict__ out_newxyz,// [B,3,S]
                                                  long long* __restrict__ out_idx) {
    const int b = blockIdx.x;
    const int t = threadIdx.x;
    const float* xb = xyz + (size_t)b * 3 * NPT;

    __shared__ float mind[NPT];       // 32 KB
    __shared__ float rd[512];
    __shared__ int   ri[512];
    __shared__ int   s_last;

    for (int n = t; n < NPT; n += 512) mind[n] = 1e10f;
    if (t == 0) s_last = 0;
    __syncthreads();

    for (int j = 0; j < SS; ++j) {
        const int last = s_last;
        const float px = xb[last];
        const float py = xb[NPT + last];
        const float pz = xb[2 * NPT + last];
        if (t == 0) {
            fps_idx[b * SS + j] = last;
            qxyz[(size_t)(b * SS + j) * 3 + 0] = px;
            qxyz[(size_t)(b * SS + j) * 3 + 1] = py;
            qxyz[(size_t)(b * SS + j) * 3 + 2] = pz;
            out_newxyz[((size_t)b * 3 + 0) * SS + j] = px;
            out_newxyz[((size_t)b * 3 + 1) * SS + j] = py;
            out_newxyz[((size_t)b * 3 + 2) * SS + j] = pz;
            out_idx[b * SS + j] = (long long)last;
        }
        float bd = -1.0f;
        int   bi = 0x7fffffff;
        for (int n = t; n < NPT; n += 512) {
            float dx = xb[n] - px, dy = xb[NPT + n] - py, dz = xb[2 * NPT + n] - pz;
            float d  = dx * dx + dy * dy + dz * dz;
            float mn = fminf(mind[n], d);
            mind[n] = mn;
            if (mn > bd) { bd = mn; bi = n; }
        }
        rd[t] = bd; ri[t] = bi;
        __syncthreads();
        for (int off = 256; off > 0; off >>= 1) {
            if (t < off) {
                float od = rd[t + off]; int oi = ri[t + off];
                if (od > rd[t] || (od == rd[t] && oi < ri[t])) { rd[t] = od; ri[t] = oi; }
            }
            __syncthreads();
        }
        if (t == 0) s_last = ri[0];
        __syncthreads();
    }
}

// ---------------------------------------------------------------------------
// 2) Brute-force KNN (K=32): one thread per query, sorted insertion list.
//    Neighbor ORDER does not affect downstream (sum over k), only membership.
// ---------------------------------------------------------------------------
__global__ __launch_bounds__(256) void knn_kernel(const float* __restrict__ xyz,  // [B,3,N]
                                                  const float* __restrict__ qxyz, // [B,S,3]
                                                  int* __restrict__ knn_idx) {    // [B,S,K]
    int q = blockIdx.x * blockDim.x + threadIdx.x;   // 0 .. B*S-1
    if (q >= NB * SS) return;
    int b = q >> 11;
    const float* xb = xyz + (size_t)b * 3 * NPT;
    const float qx = qxyz[(size_t)q * 3 + 0];
    const float qy = qxyz[(size_t)q * 3 + 1];
    const float qz = qxyz[(size_t)q * 3 + 2];

    float bd[KN];
    int   bi[KN];
#pragma unroll
    for (int i = 0; i < KN; ++i) { bd[i] = 3.0e38f; bi[i] = 0; }

    for (int n = 0; n < NPT; ++n) {
        float dx = xb[n] - qx, dy = xb[NPT + n] - qy, dz = xb[2 * NPT + n] - qz;
        float d  = dx * dx + dy * dy + dz * dz;
        if (d < bd[KN - 1]) {
            int j = KN - 1;
            while (j > 0 && bd[j - 1] > d) { bd[j] = bd[j - 1]; bi[j] = bi[j - 1]; --j; }
            bd[j] = d; bi[j] = n;
        }
    }
    int* o = knn_idx + (size_t)q * KN;
#pragma unroll
    for (int i = 0; i < KN; ++i) o[i] = bi[i];
}

// ---------------------------------------------------------------------------
// 3) Gather + WeightNet MLP + per-point agg GEMM (64x16x32) via 4x
//    v_wmma_f32_16x16x32_bf16. One wave per sampled point (lane == neighbor).
//    np stored c-major in LDS so every A-fragment is two ds_load_b128.
// ---------------------------------------------------------------------------
__global__ __launch_bounds__(256) void group_agg_kernel(
        const float* __restrict__ xyz,    // [B,3,N]
        const float* __restrict__ points, // [B,D,N]
        const float* __restrict__ qxyz,   // [B,S,3]
        const int*   __restrict__ knn_idx,// [B,S,K]
        const float* __restrict__ w1, const float* __restrict__ b1,
        const float* __restrict__ w2, const float* __restrict__ b2,
        const float* __restrict__ w3, const float* __restrict__ b3,
        __bf16* __restrict__ aggbf) {     // [B*S, 1024]  (c*16 + w)
    const int wave = threadIdx.x >> 5;
    const int lane = threadIdx.x & 31;
    const int m    = blockIdx.x * 8 + wave;   // global point id, 0..16383
    const int b    = m >> 11;

    __shared__ __bf16 np_s[8][CIN][KNP];  // 48 KB : new_points[c][k], k-padded
    __shared__ __bf16 wt_s[8][KN][WNW];   //  8 KB : weights[k][w]

    const int   idx = knn_idx[(size_t)m * KN + lane];
    const float qx  = qxyz[(size_t)m * 3 + 0];
    const float qy  = qxyz[(size_t)m * 3 + 1];
    const float qz  = qxyz[(size_t)m * 3 + 2];
    const float* xb = xyz + (size_t)b * 3 * NPT;
    const float dx = xb[idx] - qx;
    const float dy = xb[NPT + idx] - qy;
    const float dz = xb[2 * NPT + idx] - qz;

    // WeightNet: 3 -> 8 -> 8 -> 16, ReLU
    float h1[8];
#pragma unroll
    for (int i = 0; i < 8; ++i) {
        float a = w1[i * 3 + 0] * dx + w1[i * 3 + 1] * dy + w1[i * 3 + 2] * dz + b1[i];
        h1[i] = a > 0.0f ? a : 0.0f;
    }
    float h2[8];
#pragma unroll
    for (int i = 0; i < 8; ++i) {
        float a = b2[i];
#pragma unroll
        for (int j = 0; j < 8; ++j) a += w2[i * 8 + j] * h1[j];
        h2[i] = a > 0.0f ? a : 0.0f;
    }
#pragma unroll
    for (int i = 0; i < WNW; ++i) {
        float a = b3[i];
#pragma unroll
        for (int j = 0; j < 8; ++j) a += w3[i * 8 + j] * h2[j];
        wt_s[wave][lane][i] = (__bf16)(a > 0.0f ? a : 0.0f);
    }

    np_s[wave][0][lane] = (__bf16)dx;
    np_s[wave][1][lane] = (__bf16)dy;
    np_s[wave][2][lane] = (__bf16)dz;
    const float* pb = points + (size_t)b * DFE * NPT;
#pragma unroll 4
    for (int c = 0; c < DFE; ++c)
        np_s[wave][3 + c][lane] = (__bf16)pb[(size_t)c * NPT + idx];

    __syncthreads();

    // B-fragment: B[k,w], n = lane&15, element i -> k = (lane>>4)*8 + (i&7) + (i>>3)*16
    v16bf Bf;
#pragma unroll
    for (int i = 0; i < 16; ++i) Bf[i] = wt_s[wave][(lane >> 4) * 8 + (i & 7) + (i >> 3) * 16][lane & 15];

    const int row0  = lane & 15;
    const int khalf = lane >> 4;
    __bf16* ao = aggbf + (size_t)m * KDIM;

#pragma unroll
    for (int t = 0; t < 4; ++t) {              // 4 M-tiles cover C=64 channels
        const int c = t * 16 + row0;
        const __bf16* rp = &np_s[wave][c][0];
        v8bf lo = *(const v8bf*)(rp + khalf * 8);        // K = khalf*8 + 0..7
        v8bf hi = *(const v8bf*)(rp + 16 + khalf * 8);   // K = 16+khalf*8 + 0..7
        v16bf Af;
#pragma unroll
        for (int i = 0; i < 8; ++i) { Af[i] = lo[i]; Af[8 + i] = hi[i]; }
        v8f acc = {};
        acc = __builtin_amdgcn_wmma_f32_16x16x32_bf16(
                  false, Af, false, Bf, (short)0, acc, false, false);
        // D layout: lane -> N = lane&15 (=w), VGPR v -> M = v + 8*khalf
#pragma unroll
        for (int v = 0; v < 8; ++v) {
            int cd = t * 16 + v + 8 * khalf;
            ao[cd * WNW + row0] = (__bf16)acc[v];
        }
    }
}

// ---------------------------------------------------------------------------
// 4) Final linear + leaky ReLU: C[16384,128] = A[16384,1024] x B[1024,128].
//    One TDM tensor_load_to_lds pulls the block's 16x1024 bf16 A panel into
//    LDS (padded 4 dw per 256 dw to de-phase banks); 8 waves cover OC=128
//    with 32 K-steps of v_wmma_f32_16x16x32_bf16 each, A-frags = ds_load_b128.
// ---------------------------------------------------------------------------
__global__ __launch_bounds__(256) void final_gemm_kernel(
        const __bf16* __restrict__ aggbf,  // [16384,1024]
        const __bf16* __restrict__ lwT,    // [1024,128]
        const float*  __restrict__ lb,     // [128]
        float* __restrict__ out) {         // [B,128,2048]
    const int wave  = threadIdx.x >> 5;
    const int lane  = threadIdx.x & 31;
    const int m0    = blockIdx.x * 16;     // 16 consecutive points (same batch)
    const int oc0   = wave * 16;
    const int row0  = lane & 15;
    const int khalf = lane >> 4;

    __shared__ __align__(16) char As[APAD_BYTES];

    if (wave == 0) {
        unsigned long long ga = (unsigned long long)(const void*)(aggbf + (size_t)m0 * KDIM);
        unsigned lds = lds_offset_of(As);
        // D# group 0: count=1 | lds_addr | global_addr[56:0] | type=2
        u32x4 g0;
        g0[0] = 1u;
        g0[1] = lds;
        g0[2] = (unsigned)(ga & 0xffffffffu);
        g0[3] = (unsigned)((ga >> 32) & 0x1ffffffu) | (2u << 30);
        // D# group 1: data_size=2B | pad 4dw per 256dw | dims/strides/tile
        i32x8 g1;
        g1[0] = (int)((1u << 16) | (1u << 20) | (7u << 22) | (3u << 25));
        g1[1] = (int)(((unsigned)KDIM & 0xffffu) << 16);   // tensor_dim0 = 1024
        g1[2] = (int)((16384u & 0xffffu) << 16);           // tensor_dim1 = 16384
        g1[3] = (int)(((unsigned)KDIM) << 16);             // tile_dim0 = 1024
        g1[4] = 16;                                        // tile_dim1 = 16
        g1[5] = KDIM;                                      // tensor_dim0_stride
        g1[6] = 0;
        g1[7] = 0;
        i32x4 g2 = {0, 0, 0, 0};
        i32x4 g3 = {0, 0, 0, 0};
#if __clang_major__ >= 23
        i32x8 g4 = {0, 0, 0, 0, 0, 0, 0, 0};
        __builtin_amdgcn_tensor_load_to_lds(g0, g1, g2, g3, g4, 0);
#else
        __builtin_amdgcn_tensor_load_to_lds(g0, g1, g2, g3, 0);
#endif
        __builtin_amdgcn_s_wait_tensorcnt((short)0);
    }
    __syncthreads();

    v8f acc = {};
    for (int kb = 0; kb < KDIM; kb += 32) {
        // dword index within padded panel for this lane's two 16B chunks
        int x  = (kb >> 1) + khalf * 4;          // dword offset in row
        int e0 = row0 * 512 + x;
        int e1 = e0 + 8;
        const char* p0 = As + e0 * 4 + (e0 >> 8) * 16;
        const char* p1 = As + e1 * 4 + (e1 >> 8) * 16;
        v8bf lo = *(const v8bf*)p0;              // K = kb + khalf*8 + 0..7
        v8bf hi = *(const v8bf*)p1;              // K = kb + 16 + khalf*8 + 0..7
        v16bf Af;
#pragma unroll
        for (int i = 0; i < 8; ++i) { Af[i] = lo[i]; Af[8 + i] = hi[i]; }

        v16bf Bf;   // B[k = kb+khalf*8+(i&7)+(i>>3)*16, n = oc0+row0] from lwT
        const __bf16* brow = lwT + (size_t)kb * OCH + oc0 + row0;
#pragma unroll
        for (int i = 0; i < 16; ++i) Bf[i] = brow[(size_t)(khalf * 8 + (i & 7) + (i >> 3) * 16) * OCH];

        acc = __builtin_amdgcn_wmma_f32_16x16x32_bf16(
                  false, Af, false, Bf, (short)0, acc, false, false);
    }

    const int oc = oc0 + row0;             // D: lane -> N (=oc)
    const float bias = lb[oc];
#pragma unroll
    for (int v = 0; v < 8; ++v) {          // D: VGPR v -> M row = v + 8*khalf
        int mm = m0 + v + 8 * khalf;
        float val = acc[v] + bias;
        val = val >= 0.0f ? val : 0.1f * val;
        int bb = mm >> 11;
        int ss = mm & (SS - 1);
        out[((size_t)bb * OCH + oc) * SS + ss] = val;
    }
}

// ---------------------------------------------------------------------------
extern "C" void kernel_launch(void* const* d_in, const int* in_sizes, int n_in,
                              void* d_out, int out_size, void* d_ws, size_t ws_size,
                              hipStream_t stream) {
    const float* xyz    = (const float*)d_in[0];
    const float* points = (const float*)d_in[1];
    const float* w1     = (const float*)d_in[2];
    const float* b1     = (const float*)d_in[3];
    const float* w2     = (const float*)d_in[4];
    const float* b2     = (const float*)d_in[5];
    const float* w3     = (const float*)d_in[6];
    const float* b3     = (const float*)d_in[7];
    const float* lw     = (const float*)d_in[8];
    const float* lb     = (const float*)d_in[9];

    // workspace layout (bytes)
    char* ws = (char*)d_ws;
    int*     fps_idx = (int*)(ws + 0);                 //   65536
    float*   qxyz    = (float*)(ws + 65536);           //  196608
    int*     knn     = (int*)(ws + 262144);            // 2097152
    __bf16*  lwT     = (__bf16*)(ws + 2359296);        //  262144
    __bf16*  aggbf   = (__bf16*)(ws + 2621440);        // 33554432  (end 36175872)

    // output: new_xyz [8,3,2048] f32 | out [8,128,2048] f32 | fps_idx [8,2048] i64
    float*     outf       = (float*)d_out;
    float*     out_newxyz = outf;
    float*     out_main   = outf + (size_t)NB * 3 * SS;
    long long* out_idx    = (long long*)(outf + (size_t)NB * 3 * SS + (size_t)NB * OCH * SS);

    prep_lw_kernel   <<<512, 256, 0, stream>>>(lw, lwT);
    fps_kernel       <<<NB, 512, 0, stream>>>(xyz, fps_idx, qxyz, out_newxyz, out_idx);
    knn_kernel       <<<(NB * SS) / 256, 256, 0, stream>>>(xyz, qxyz, knn);
    group_agg_kernel <<<(NB * SS) / 8, 256, 0, stream>>>(xyz, points, qxyz, knn,
                                                         w1, b1, w2, b2, w3, b3, aggbf);
    final_gemm_kernel<<<(NB * SS) / 16, 256, 0, stream>>>(aggbf, lwT, lb, out_main);
}
